// Attention_3513283248714
// MI455X (gfx1250) — matrix operand order repaired and checked by reading, not compiled, not run on hardware
//
#include <hip/hip_runtime.h>
#include <hip/hip_bf16.h>

// ---------------------------------------------------------------------------
// Fused attention for MI455X (gfx1250, wave32, WMMA + async LDS DMA).
// All GEMMs run on v_wmma_f32_16x16x32_bf16 with fp32 accumulation.
// ---------------------------------------------------------------------------

typedef __bf16 bf16_t;
typedef __attribute__((ext_vector_type(16))) __bf16 v16bf;
typedef __attribute__((ext_vector_type(8)))  __bf16 v8bf;
typedef __attribute__((ext_vector_type(8)))  float  v8f;

#define B_     2
#define N_     2048
#define DIM_   2048
#define NH_    16
#define HD_    128
#define E3_    6144        // 3*NH_*HD_
#define RROWS_ 4096        // B_*N_
#define INNER_ 2048        // NH_*HD_
#define EPS_   1e-5f

// ---------------------------------------------------------------------------
// Fragment packing helpers.
// A-fragment (16x32 bf16): lane<16 holds row (lane&15), K=0..15;
// lane>=16 holds same row set, K=16..31. Two contiguous 16B loads.
// B-fragment (32x16 bf16): lane holds column (lane&15); lanes<16 K=0..15,
// lanes>=16 K=16..31 — one contiguous 32B run per lane.
// ---------------------------------------------------------------------------
__device__ __forceinline__ v16bf pack16(const bf16_t* p, int hi_off) {
  v8bf lo = *(const v8bf*)(p);
  v8bf hi = *(const v8bf*)(p + hi_off);
  v16bf r;
#pragma unroll
  for (int i = 0; i < 8; ++i) { r[i] = lo[i]; r[i + 8] = hi[i]; }
  return r;
}

__device__ __forceinline__ v8f wmma_bf16(v16bf a, v16bf b, v8f c) {
  return __builtin_amdgcn_wmma_f32_16x16x32_bf16(
      /*neg_a=*/false, a, /*neg_b=*/false, b,
      /*c_mod=*/(short)0, c, /*reuse_a=*/false, /*reuse_b=*/false);
}

__device__ __forceinline__ float rmax16(float v) {
#pragma unroll
  for (int m = 1; m < 16; m <<= 1) v = fmaxf(v, __shfl_xor(v, m, 32));
  return v;
}
__device__ __forceinline__ float rsum16(float v) {
#pragma unroll
  for (int m = 1; m < 16; m <<= 1) v += __shfl_xor(v, m, 32);
  return v;
}

// gfx1250 async global->LDS copy, 16B per active lane. ASYNCcnt-tracked.
// %0 = LDS byte address (low 32 bits of flat shared pointer), %1 = global addr.
__device__ __forceinline__ void async_copy16(unsigned lds_byte,
                                             unsigned long long gaddr) {
  asm volatile("global_load_async_to_lds_b128 %0, %1, off"
               :: "v"(lds_byte), "v"(gaddr) : "memory");
}
__device__ __forceinline__ void wait_async_le8() {
  asm volatile("s_wait_asynccnt 0x8" ::: "memory");
}
__device__ __forceinline__ void wait_async_0() {
  asm volatile("s_wait_asynccnt 0x0" ::: "memory");
}

// ---------------------------------------------------------------------------
// Kernel: fp32 -> bf16 convert (grid-stride)
// ---------------------------------------------------------------------------
__global__ void cvt_f32_bf16(const float* __restrict__ in,
                             bf16_t* __restrict__ out, int n) {
  int i = blockIdx.x * blockDim.x + threadIdx.x;
  int stride = gridDim.x * blockDim.x;
  for (; i < n; i += stride) out[i] = (bf16_t)in[i];
}

// ---------------------------------------------------------------------------
// Kernel: bf16 GEMM, C(f32)[M,Nn] = A[M,K] * Bw^T  where Bw is [Nn,K]
// (Bw rows are the K-contiguous columns of the B operand).
// One wave per block; wave tile = 64(M) x 64(N); K stepped by 32.
// 16 WMMAs per k-step @ 256B/lane loads -> ~32 flops/byte.
// ---------------------------------------------------------------------------
__global__ __launch_bounds__(32)
void gemm_bf16(const bf16_t* __restrict__ A, const bf16_t* __restrict__ Bw,
               float* __restrict__ C, int M, int Nn, int K) {
  const int lane = threadIdx.x & 31;
  const int half = lane >> 4;
  const int l16  = lane & 15;
  const int n0 = blockIdx.x * 64;
  const int m0 = blockIdx.y * 64;

  v8f acc[4][4];
#pragma unroll
  for (int t = 0; t < 4; ++t)
#pragma unroll
    for (int u = 0; u < 4; ++u)
#pragma unroll
      for (int j = 0; j < 8; ++j) acc[t][u][j] = 0.f;

  const bf16_t* a_base = A + (size_t)(m0 + l16) * K + half * 16;
  const bf16_t* b_base = Bw + (size_t)(n0 + l16) * K + half * 16;

  for (int kk = 0; kk < K; kk += 32) {
    v16bf a[4];
#pragma unroll
    for (int t = 0; t < 4; ++t) {
      const bf16_t* ap = a_base + (size_t)(t * 16) * K + kk;
      __builtin_prefetch((const void*)(ap + 32), 0, 3);  // next k-step, WGP scope
      a[t] = pack16(ap, 8);
    }
    v16bf b[4];
#pragma unroll
    for (int u = 0; u < 4; ++u) {
      const bf16_t* bp = b_base + (size_t)(u * 16) * K + kk;
      __builtin_prefetch((const void*)(bp + 32), 0, 3);
      b[u] = pack16(bp, 8);
    }
#pragma unroll
    for (int t = 0; t < 4; ++t)
#pragma unroll
      for (int u = 0; u < 4; ++u)
        acc[t][u] = wmma_bf16(a[t], b[u], acc[t][u]);
  }

#pragma unroll
  for (int t = 0; t < 4; ++t)
#pragma unroll
    for (int u = 0; u < 4; ++u)
#pragma unroll
      for (int j = 0; j < 8; ++j) {
        int row = m0 + t * 16 + j + half * 8;
        int col = n0 + u * 16 + l16;
        C[(size_t)row * Nn + col] = acc[t][u][j];
      }
}

// ---------------------------------------------------------------------------
// Kernel: RMSNorm + RoPE + layout transform.
// qkv f32 [B*N, 3, NH, HD]  ->  q,k bf16 [B,NH,N,HD] (1/sqrt(HD) folded in q)
//                               vT  bf16 [B,NH,HD,N]
// One block per (b,n,h), 128 threads (one per hd).
// ---------------------------------------------------------------------------
__global__ __launch_bounds__(128)
void qkv_transform(const float* __restrict__ qkv, const float* __restrict__ freqs,
                   const float* __restrict__ qg, const float* __restrict__ kg,
                   bf16_t* __restrict__ qb, bf16_t* __restrict__ kb,
                   bf16_t* __restrict__ vtb) {
  const int idx = blockIdx.x;
  const int h  = idx & (NH_ - 1);
  const int bn = idx >> 4;              // b*N_ + n
  const int n  = bn & (N_ - 1);
  const int b  = bn >> 11;
  const int hd = threadIdx.x;

  const size_t base = (size_t)bn * E3_ + (size_t)h * HD_;
  float qv = qkv[base + 0 * INNER_ + hd];
  float kv = qkv[base + 1 * INNER_ + hd];
  float vv = qkv[base + 2 * INNER_ + hd];

  __shared__ float red[HD_];
  __shared__ float qs[HD_];
  __shared__ float ks[HD_];

  red[hd] = qv * qv;
  __syncthreads();
#pragma unroll
  for (int s = 64; s > 0; s >>= 1) {
    if (hd < s) red[hd] += red[hd + s];
    __syncthreads();
  }
  float rq = rsqrtf(red[0] * (1.0f / HD_) + EPS_);
  __syncthreads();

  red[hd] = kv * kv;
  __syncthreads();
#pragma unroll
  for (int s = 64; s > 0; s >>= 1) {
    if (hd < s) red[hd] += red[hd + s];
    __syncthreads();
  }
  float rk = rsqrtf(red[0] * (1.0f / HD_) + EPS_);
  __syncthreads();

  const float qscale = 0.08838834764831845f;  // 1/sqrt(HD)
  qs[hd] = qv * rq * qg[hd] * qscale;
  ks[hd] = kv * rk * kg[hd];
  __syncthreads();

  const int i = hd >> 1;
  float c = freqs[((size_t)bn * (HD_ / 2) + i) * 2 + 0];
  float s = freqs[((size_t)bn * (HD_ / 2) + i) * 2 + 1];
  float q0 = qs[2 * i], q1 = qs[2 * i + 1];
  float k0 = ks[2 * i], k1 = ks[2 * i + 1];
  float qo = (hd & 1) ? (q0 * s + q1 * c) : (q0 * c - q1 * s);
  float ko = (hd & 1) ? (k0 * s + k1 * c) : (k0 * c - k1 * s);

  const size_t o = ((size_t)(b * NH_ + h) * N_ + n) * HD_ + hd;
  qb[o] = (bf16_t)qo;
  kb[o] = (bf16_t)ko;
  vtb[((size_t)(b * NH_ + h) * HD_ + hd) * N_ + n] = (bf16_t)vv;
}

// ---------------------------------------------------------------------------
// Kernel: flash attention. 4-wave workgroup covers 64 query rows of one
// (b,h); each wave owns 16 rows. Keys streamed in 32-wide chunks with
// DOUBLE-BUFFERED async global->LDS DMA (global_load_async_to_lds_b128):
// while computing on tile `buf`, chunk+1 streams into `buf^1`; the top-of-
// iteration s_wait_asynccnt 8 only drains the older 8 ops (async loads
// complete in order), hiding L2 latency behind 16 WMMAs + softmax.
//   - S = Q*K^T : 8 WMMAs/wave/chunk from LDS fragments
//   - online softmax in fp32 (C-layout lane group == row half)
//   - P converts C-layout -> A-layout through per-wave LDS tile
//   - O += P*V : 8 WMMAs/wave/chunk
// Output written bf16 to aout[B,N,NH*HD] for the final projection GEMM.
// ---------------------------------------------------------------------------
__global__ __launch_bounds__(128)
void attn_fwd(const bf16_t* __restrict__ qb, const bf16_t* __restrict__ kb,
              const bf16_t* __restrict__ vtb, bf16_t* __restrict__ aout) {
  const int bh   = blockIdx.y;            // b*NH_ + h
  const int b = bh >> 4, h = bh & 15;
  const int tid  = threadIdx.x;
  const int wave = tid >> 5;
  const int lane = tid & 31;
  const int half = lane >> 4;
  const int l16  = lane & 15;
  const int qrow0 = blockIdx.x * 64 + wave * 16;

  __shared__ __align__(16) bf16_t kt[2][32][HD_];     // 16KB key tiles (row=key)
  __shared__ __align__(16) bf16_t vt[2][HD_][32];     // 16KB v^T tiles (row=hd)
  __shared__ __align__(16) bf16_t pl[4][16][32];      // 4KB  per-wave P tiles

  const unsigned kt_lds = (unsigned)(unsigned long long)(&kt[0][0][0]);
  const unsigned vt_lds = (unsigned)(unsigned long long)(&vt[0][0][0]);
  const unsigned long long kg_base =
      (unsigned long long)(const void*)(kb + (size_t)bh * N_ * HD_);
  const unsigned long long vg_base =
      (unsigned long long)(const void*)(vtb + ((size_t)bh * HD_ + tid) * N_);

  // Issue one chunk's staging: 8 async ops/wave (4 K-tile + 4 vT-tile).
  // K tile is one contiguous 8KB block; vT tile rows are 64B each.
  auto issue_stage = [&](int buf, int kc) {
    unsigned      kl = kt_lds + buf * 8192u + tid * 16u;
    unsigned long long kg = kg_base + (size_t)kc * HD_ * 2 + tid * 16u;
#pragma unroll
    for (int i = 0; i < 4; ++i)
      async_copy16(kl + i * 2048u, kg + (unsigned long long)i * 2048u);
    unsigned      vl = vt_lds + buf * 8192u + tid * 64u;
    unsigned long long vg = vg_base + (size_t)kc * 2;
#pragma unroll
    for (int i = 0; i < 4; ++i)
      async_copy16(vl + i * 16u, vg + (unsigned long long)i * 16u);
  };

  // Q A-fragments (16 x 128 over 4 K-steps), resident for the whole loop
  const bf16_t* qp = qb + ((size_t)bh * N_ + qrow0) * HD_;
  v16bf aq[4];
#pragma unroll
  for (int t = 0; t < 4; ++t)
    aq[t] = pack16(qp + (size_t)l16 * HD_ + t * 32 + half * 16, 8);

  float mrun[8], lrun[8];
  v8f oacc[8];
#pragma unroll
  for (int j = 0; j < 8; ++j) { mrun[j] = -1e30f; lrun[j] = 0.f; }
#pragma unroll
  for (int u = 0; u < 8; ++u)
#pragma unroll
    for (int j = 0; j < 8; ++j) oacc[u][j] = 0.f;

  issue_stage(0, 0);

  const int nchunks = N_ / 32;
  for (int it = 0; it < nchunks; ++it) {
    const int buf = it & 1;
    if (it + 1 < nchunks) {
      issue_stage(buf ^ 1, (it + 1) * 32);   // prefetch next chunk into other buf
      wait_async_le8();                      // drain only the older 8 ops
    } else {
      wait_async_0();
    }
    __syncthreads();                         // tiles visible to all 4 waves

    // ---- S = Q * K^T  (two 16-col key groups) from LDS ----
    v8f s0, s1;
#pragma unroll
    for (int j = 0; j < 8; ++j) { s0[j] = 0.f; s1[j] = 0.f; }
#pragma unroll
    for (int t = 0; t < 4; ++t) {
      s0 = wmma_bf16(aq[t], pack16(&kt[buf][l16][t * 32 + half * 16], 8), s0);
      s1 = wmma_bf16(aq[t], pack16(&kt[buf][16 + l16][t * 32 + half * 16], 8), s1);
    }

    // ---- online softmax (C layout: lane group == row half) ----
    float sc[8];
#pragma unroll
    for (int j = 0; j < 8; ++j) {
      float rm = rmax16(fmaxf(s0[j], s1[j]));
      float mn = fmaxf(mrun[j], rm);
      sc[j] = __expf(mrun[j] - mn);
      mrun[j] = mn;
      float p0 = __expf(s0[j] - mn);
      float p1 = __expf(s1[j] - mn);
      s0[j] = p0; s1[j] = p1;
      lrun[j] = lrun[j] * sc[j] + rsum16(p0 + p1);
    }
#pragma unroll
    for (int u = 0; u < 8; ++u)
#pragma unroll
      for (int j = 0; j < 8; ++j) oacc[u][j] *= sc[j];

    // ---- P: C layout -> per-wave LDS tile -> A layout ----
#pragma unroll
    for (int j = 0; j < 8; ++j) {
      pl[wave][j + half * 8][l16]      = (bf16_t)s0[j];
      pl[wave][j + half * 8][l16 + 16] = (bf16_t)s1[j];
    }
    __syncthreads();
    v16bf pa = pack16(&pl[wave][l16][half * 16], 8);

    // ---- O += P * V  (vT rows from LDS) ----
#pragma unroll
    for (int u = 0; u < 8; ++u)
      oacc[u] = wmma_bf16(pa, pack16(&vt[buf][u * 16 + l16][half * 16], 8),
                          oacc[u]);
    __syncthreads();   // all reads of this buf done before it is re-staged
  }

  // ---- finalize: divide by row sums, store bf16 [B,N,NH*HD] ----
  float linv[8];
#pragma unroll
  for (int j = 0; j < 8; ++j) linv[j] = 1.0f / lrun[j];
#pragma unroll
  for (int u = 0; u < 8; ++u)
#pragma unroll
    for (int j = 0; j < 8; ++j) {
      int row = qrow0 + j + half * 8;
      int col = h * HD_ + u * 16 + l16;
      aout[((size_t)b * N_ + row) * INNER_ + col] = (bf16_t)(oacc[u][j] * linv[j]);
    }
}

// ---------------------------------------------------------------------------
// Host-side orchestration
// ---------------------------------------------------------------------------
extern "C" void kernel_launch(void* const* d_in, const int* in_sizes, int n_in,
                              void* d_out, int out_size, void* d_ws, size_t ws_size,
                              hipStream_t stream) {
  (void)in_sizes; (void)n_in; (void)out_size; (void)ws_size;
  const float* x     = (const float*)d_in[0];   // [B,N,DIM]
  const float* Wqkv  = (const float*)d_in[1];   // [E3,DIM]
  const float* qg    = (const float*)d_in[2];   // [HD]
  const float* kg    = (const float*)d_in[3];   // [HD]
  const float* Wout  = (const float*)d_in[4];   // [DIM,INNER]
  const float* freqs = (const float*)d_in[5];   // [B,N,HD/2,2]
  float* out = (float*)d_out;                   // [B,N,DIM]

  char* ws = (char*)d_ws;
  size_t off = 0;
  bf16_t* x_bf    = (bf16_t*)(ws + off); off += (size_t)RROWS_ * DIM_ * 2;
  bf16_t* wqkv_bf = (bf16_t*)(ws + off); off += (size_t)E3_ * DIM_ * 2;
  bf16_t* wout_bf = (bf16_t*)(ws + off); off += (size_t)DIM_ * INNER_ * 2;
  float*  qkv     = (float*)(ws + off);  off += (size_t)RROWS_ * E3_ * 4;
  bf16_t* q_bf    = (bf16_t*)(ws + off); off += (size_t)B_ * NH_ * N_ * HD_ * 2;
  bf16_t* k_bf    = (bf16_t*)(ws + off); off += (size_t)B_ * NH_ * N_ * HD_ * 2;
  bf16_t* vt_bf   = (bf16_t*)(ws + off); off += (size_t)B_ * NH_ * N_ * HD_ * 2;
  bf16_t* aout_bf = (bf16_t*)qkv;  // qkv consumed by transform; reuse for attn out

  // 1) fp32 -> bf16 conversions
  {
    int nx = RROWS_ * DIM_;
    cvt_f32_bf16<<<(nx + 255) / 256, 256, 0, stream>>>(x, x_bf, nx);
    int nw = E3_ * DIM_;
    cvt_f32_bf16<<<(nw + 255) / 256, 256, 0, stream>>>(Wqkv, wqkv_bf, nw);
    int no = DIM_ * INNER_;
    cvt_f32_bf16<<<(no + 255) / 256, 256, 0, stream>>>(Wout, wout_bf, no);
  }

  // 2) QKV projection: [4096,2048] x [2048,6144] -> f32 qkv
  gemm_bf16<<<dim3(E3_ / 64, RROWS_ / 64), 32, 0, stream>>>(
      x_bf, wqkv_bf, qkv, RROWS_, E3_, DIM_);

  // 3) RMSNorm + RoPE + layout transform
  qkv_transform<<<B_ * N_ * NH_, 128, 0, stream>>>(
      qkv, freqs, qg, kg, q_bf, k_bf, vt_bf);

  // 4) attention (flash, online softmax, double-buffered async LDS tiles)
  attn_fwd<<<dim3(N_ / 64, B_ * NH_), 128, 0, stream>>>(q_bf, k_bf, vt_bf, aout_bf);

  // 5) output projection: [4096,2048] x [2048,2048] -> f32 out
  gemm_bf16<<<dim3(DIM_ / 64, RROWS_ / 64), 32, 0, stream>>>(
      aout_bf, wout_bf, out, RROWS_, DIM_, DIM_);
}
